// SingleHeadAttention_11751030522720
// MI455X (gfx1250) — matrix operand order repaired and checked by reading, not compiled, not run on hardware
//
#include <hip/hip_runtime.h>
#include <cstdint>
#include <cstddef>

typedef unsigned short u16;
typedef __attribute__((ext_vector_type(16))) __bf16 v16bf;
typedef __attribute__((ext_vector_type(8)))  float  v8f;

// ---------- small helpers ----------
__device__ __forceinline__ u16 f2bf(float x) {           // fp32 -> bf16, RNE
  unsigned int u = __float_as_uint(x);
  u += 0x7FFFu + ((u >> 16) & 1u);
  return (u16)(u >> 16);
}
__device__ __forceinline__ float bf2f(u16 h) {
  return __uint_as_float(((unsigned int)h) << 16);
}
__device__ __forceinline__ unsigned int pk2(float a, float b) {
  return (unsigned int)f2bf(a) | ((unsigned int)f2bf(b) << 16);
}

__device__ __forceinline__ v8f wmma_bf16(v16bf a, v16bf b, v8f c) {
  // (neg_a, A, neg_b, B, c_mod, C, reuse_a, reuse_b)
  return __builtin_amdgcn_wmma_f32_16x16x32_bf16(false, a, false, b, (short)0, c,
                                                 false, false);
}

// ---------- CDNA5 async global->LDS copy (ASYNCcnt path, no VGPR data) ----------
// Generic pointers into LDS carry the LDS byte address in their low 32 bits
// (flat-to-LDS translation truncates, ISA 10.2), so we can feed VDST directly.
__device__ __forceinline__ unsigned lds_off(const void* p) {
  return (unsigned)(uintptr_t)p;
}
__device__ __forceinline__ void async_ld_b128(unsigned dst, const void* g) {
  asm volatile("global_load_async_to_lds_b128 %0, %1, off"
               :: "v"(dst), "v"((unsigned long long)(uintptr_t)g)
               : "memory");
}
__device__ __forceinline__ void async_ld_b64(unsigned dst, const void* g) {
  asm volatile("global_load_async_to_lds_b64 %0, %1, off"
               :: "v"(dst), "v"((unsigned long long)(uintptr_t)g)
               : "memory");
}
template <int N>
__device__ __forceinline__ void wait_async() {
#if __has_builtin(__builtin_amdgcn_s_wait_asynccnt)
  __builtin_amdgcn_s_wait_asynccnt((short)N);
#else
  asm volatile("s_wait_asynccnt %0" :: "n"(N) : "memory");
#endif
}

// Load one 16x32 bf16 WMMA operand (A and B use the same row-per-lane striping):
// lanes 0-15 hold K 0..7 / 16..23 of row (lane&15); lanes 16-31 hold K 8..15 / 24..31.
struct Frag32B { uint4 lo; uint4 hi; };
union FragU { Frag32B u; v16bf v; };
__device__ __forceinline__ v16bf frag_lds(const u16* base, int pitch, int lane) {
  const int row = lane & 15;
  const int h8  = (lane >> 4) << 3;
  const u16* p = base + row * pitch + h8;
  FragU f;
  f.u.lo = *(const uint4*)(p);
  f.u.hi = *(const uint4*)(p + 16);
  return f.v;
}

// ---------- kernel 0: fp32 -> bf16 conversion ----------
__global__ void cvt_f32_bf16(const float* __restrict__ in, u16* __restrict__ out,
                             int n4) {
  int i = blockIdx.x * blockDim.x + threadIdx.x;
  if (i >= n4) return;
  float4 f = ((const float4*)in)[i];
  uint2 o;
  o.x = pk2(f.x, f.y);
  o.y = pk2(f.z, f.w);
  ((uint2*)out)[i] = o;
}

// ---------- kernel 1: fused QKV projection GEMM ----------
// C[m,n] = sum_e X[m,e] * W[n,e]; grid.z = {Q,K,V}; V stored transposed.
// Double-buffered: stage k+1 via async-to-LDS while WMMAs consume stage k.
#define PKC 64
#define PP  72                      // LDS row pitch (u16): 144B, 16B-mult, conflict-free
#define PROJ_STRIDE ((64 + 128) * PP)  // one stage: Xs(64xPP)+Ws(128xPP)

__global__ __launch_bounds__(256)
void proj_gemm(const u16* __restrict__ X,
               const u16* __restrict__ Wq, const u16* __restrict__ Wk,
               const u16* __restrict__ Wv,
               u16* __restrict__ Qb, u16* __restrict__ Kb, u16* __restrict__ Vt) {
  extern __shared__ u16 dynsmem[];
  const int z  = blockIdx.z;
  const u16* W = (z == 0) ? Wq : (z == 1) ? Wk : Wv;
  const int m0 = blockIdx.y * 64;
  const int n0 = blockIdx.x * 128;
  const int tid  = threadIdx.x;
  const int lane = tid & 31;
  const int wave = tid >> 5;       // 8 waves: 2(M) x 4(N), 32x32 per wave
  const int wm = wave & 1;
  const int wn = wave >> 1;

  // 6 async ops per thread per stage (2 for X strip, 4 for W strip)
  auto stage = [&](int k0, u16* base) {
    u16* Xs = base;
    u16* Ws = base + 64 * PP;
    {   // X strip 64x64
      const int r = tid >> 2, c = (tid & 3) << 4;
      const u16* g = X + (size_t)(m0 + r) * 1024 + k0 + c;
      const unsigned o = lds_off(Xs + r * PP + c);
      async_ld_b128(o, g);
      async_ld_b128(o + 16, g + 8);
    }
    {   // W strip 128x64
      const int r = tid >> 1, c = (tid & 1) << 5;
      const u16* g = W + (size_t)(n0 + r) * 1024 + k0 + c;
      const unsigned o = lds_off(Ws + r * PP + c);
      async_ld_b128(o,      g);
      async_ld_b128(o + 16, g + 8);
      async_ld_b128(o + 32, g + 16);
      async_ld_b128(o + 48, g + 24);
    }
  };

  v8f acc[2][2] = {};
  stage(0, dynsmem);
  int cur = 0;
  for (int k0 = 0; k0 < 1024; k0 += PKC) {
    if (k0 + PKC < 1024) {
      stage(k0 + PKC, dynsmem + (cur ^ 1) * PROJ_STRIDE);  // prefetch next stage
      wait_async<6>();                // previous stage's 6 ops done (in-order)
    } else {
      wait_async<0>();
    }
    __syncthreads();
    const u16* Xs = dynsmem + cur * PROJ_STRIDE;
    const u16* Ws = Xs + 64 * PP;
#pragma unroll
    for (int kk = 0; kk < PKC; kk += 32) {
      v16bf a0 = frag_lds(Xs + (wm * 32 +  0) * PP + kk, PP, lane);
      v16bf a1 = frag_lds(Xs + (wm * 32 + 16) * PP + kk, PP, lane);
      v16bf b0 = frag_lds(Ws + (wn * 32 +  0) * PP + kk, PP, lane);
      v16bf b1 = frag_lds(Ws + (wn * 32 + 16) * PP + kk, PP, lane);
      acc[0][0] = wmma_bf16(a0, b0, acc[0][0]);
      acc[0][1] = wmma_bf16(a0, b1, acc[0][1]);
      acc[1][0] = wmma_bf16(a1, b0, acc[1][0]);
      acc[1][1] = wmma_bf16(a1, b1, acc[1][1]);
    }
    __syncthreads();    // all waves done with this buffer before it is re-staged
    cur ^= 1;
  }
  // C layout: VGPR r -> row (r + 8*(lane>=16)), col (lane&15)
  const int hi = (lane >> 4) << 3;
  const int nl = lane & 15;
#pragma unroll
  for (int i = 0; i < 2; ++i) {
    const int mt = m0 + wm * 32 + i * 16 + hi;       // 8 consecutive token rows
#pragma unroll
    for (int j = 0; j < 2; ++j) {
      const int nt = n0 + wn * 32 + j * 16 + nl;
      if (z == 2) {
        // Vt[b][n][s]: 8 consecutive tokens -> one contiguous b128 store
        const int bb = mt >> 11, s = mt & 2047;
        uint4 pk;
        pk.x = pk2(acc[i][j][0], acc[i][j][1]);
        pk.y = pk2(acc[i][j][2], acc[i][j][3]);
        pk.z = pk2(acc[i][j][4], acc[i][j][5]);
        pk.w = pk2(acc[i][j][6], acc[i][j][7]);
        *(uint4*)(Vt + ((((size_t)bb << 10) + nt) << 11) + s) = pk;
      } else {
        u16* dst = (z == 0) ? Qb : Kb;
#pragma unroll
        for (int r = 0; r < 8; ++r)
          dst[(size_t)(mt + r) * 1024 + nt] = f2bf(acc[i][j][r]);
      }
    }
  }
}

// ---------- kernel 2: causal scores + softmax (unnormalized exp, L saved) ----------
#define QP 1032  // 1024 + 8 pad: 2064B row stride, 16B aligned, conflict-free

__global__ __launch_bounds__(64)
void attn_scores(const u16* __restrict__ Qb, const u16* __restrict__ Kb,
                 u16* __restrict__ P, float* __restrict__ Lrow) {
  extern __shared__ u16 dynsmem[];
  u16*   Qs   = dynsmem;                      // 32 x QP, Q tile resident in LDS
  u16*   Ks   = dynsmem + 32 * QP;            // 32 x QP
  float* mrow = (float*)(dynsmem + 64 * QP);  // 32 row maxima
  const int b   = blockIdx.y;
  const int q0  = blockIdx.x * 32;
  const int tid  = threadIdx.x;
  const int lane = tid & 31;
  const int wave = tid >> 5;                  // 2 waves x 16 query rows
  const int hi = (lane >> 4) << 3;
  const int nl = lane & 15;
  {   // stage Q tile 32x1024 (64 KB) once via async-to-LDS
    const u16* g = Qb + (size_t)(b * 2048 + q0) * 1024;
    const int r  = tid >> 1;
    const int cb = (tid & 1) * 512;
    for (int it = 0; it < 64; ++it) {
      const int c = cb + it * 8;
      async_ld_b128(lds_off(Qs + r * QP + c), g + r * 1024 + c);
    }
  }
  wait_async<0>();
  __syncthreads();
  float mrun[8];
#pragma unroll
  for (int r = 0; r < 8; ++r) mrun[r] = -3.0e38f;
  const float scale = 0.03125f;               // 1/sqrt(1024)
  u16* Pb = P + ((size_t)b << 22);            // b * 2048 * 2048
  const int kmax = q0 + 32;                   // causal: keys [0, q0+32)
  for (int kb = 0; kb < kmax; kb += 32) {
    {   // stage K block 32x1024 via async-to-LDS
      const u16* g = Kb + (size_t)(b * 2048 + kb) * 1024;
      const int r  = tid >> 1;
      const int cb = (tid & 1) * 512;
      for (int it = 0; it < 64; ++it) {
        const int c = cb + it * 8;
        async_ld_b128(lds_off(Ks + r * QP + c), g + r * 1024 + c);
      }
    }
    wait_async<0>();
    __syncthreads();
    v8f s0 = {}, s1 = {};
    const u16* qb2 = Qs + (wave * 16) * QP;
    for (int e = 0; e < 1024; e += 32) {
      v16bf a  = frag_lds(qb2 + e, QP, lane);
      v16bf b0 = frag_lds(Ks + e, QP, lane);
      v16bf b1 = frag_lds(Ks + 16 * QP + e, QP, lane);
      s0 = wmma_bf16(a, b0, s0);
      s1 = wmma_bf16(a, b1, s1);
    }
    __syncthreads();
    const int qrow = q0 + wave * 16 + hi;
#pragma unroll
    for (int r = 0; r < 8; ++r) {
      const int q = qrow + r;
      float v0 = s0[r] * scale;
      float v1 = s1[r] * scale;
      if (kb + nl > q)      v0 = -3.0e38f;    // causal mask -> exp() == 0
      if (kb + 16 + nl > q) v1 = -3.0e38f;
      float mv = fmaxf(v0, v1);               // row-max across 16 lanes
      mv = fmaxf(mv, __shfl_xor(mv, 1));
      mv = fmaxf(mv, __shfl_xor(mv, 2));
      mv = fmaxf(mv, __shfl_xor(mv, 4));
      mv = fmaxf(mv, __shfl_xor(mv, 8));
      mrun[r] = fmaxf(mrun[r], mv);
      Pb[(size_t)q * 2048 + kb + nl]      = f2bf(v0);
      Pb[(size_t)q * 2048 + kb + 16 + nl] = f2bf(v1);
    }
  }
  if (nl == 0) {
#pragma unroll
    for (int r = 0; r < 8; ++r) mrow[wave * 16 + hi + r] = mrun[r];
  }
  __threadfence();          // make our P stores visible before re-reading
  __syncthreads();
  if (tid < 32) {           // one thread per query row: exp + row sum
    const int q   = q0 + tid;
    const float m = mrow[tid];
    float l = 0.f;
    u16* prow = Pb + (size_t)q * 2048;
    for (int k = 0; k < kmax; k += 8) {
      uint4 u = *(uint4*)(prow + k);
      float e0 = __expf(bf2f((u16)(u.x & 0xFFFFu)) - m);
      float e1 = __expf(bf2f((u16)(u.x >> 16)) - m);
      float e2 = __expf(bf2f((u16)(u.y & 0xFFFFu)) - m);
      float e3 = __expf(bf2f((u16)(u.y >> 16)) - m);
      float e4 = __expf(bf2f((u16)(u.z & 0xFFFFu)) - m);
      float e5 = __expf(bf2f((u16)(u.z >> 16)) - m);
      float e6 = __expf(bf2f((u16)(u.w & 0xFFFFu)) - m);
      float e7 = __expf(bf2f((u16)(u.w >> 16)) - m);
      l += ((e0 + e1) + (e2 + e3)) + ((e4 + e5) + (e6 + e7));
      u.x = pk2(e0, e1); u.y = pk2(e2, e3);
      u.z = pk2(e4, e5); u.w = pk2(e6, e7);
      *(uint4*)(prow + k) = u;
    }
    Lrow[b * 2048 + q] = l;
  }
}

// ---------- kernel 3: O = (P @ V) / L ----------
#define GP 56                          // 112B row stride: 16B-mult, conflict-free
#define OUT_STRIDE ((32 + 128) * GP)   // one stage: As(32xGP)+Bs(128xGP)

__global__ __launch_bounds__(256)
void attn_out(const u16* __restrict__ P, const u16* __restrict__ Vt,
              const float* __restrict__ Lrow, float* __restrict__ Out) {
  extern __shared__ u16 dynsmem[];
  const int b  = blockIdx.z;
  const int m0 = blockIdx.y * 32;
  const int n0 = blockIdx.x * 128;
  const int tid  = threadIdx.x;
  const int lane = tid & 31;
  const int wave = tid >> 5;     // 8 waves: 2(M) x 4(N), 16x32 per wave
  const int wm = wave & 1, wn = wave >> 1;
  const u16* Pb = P  + ((size_t)b << 22);   // 2048*2048
  const u16* Vb = Vt + ((size_t)b << 21);   // 1024*2048

  // 3 async ops per thread per stage
  auto stage = [&](int k0, u16* base) {
    u16* As = base;
    u16* Bs = base + 32 * GP;
    {   // P strip 32x32
      const int r = tid >> 3, c = (tid & 7) << 2;
      async_ld_b64(lds_off(As + r * GP + c),
                   Pb + (size_t)(m0 + r) * 2048 + k0 + c);
    }
    {   // Vt strip 128x32
      const int r = tid >> 1, c = (tid & 1) << 4;
      async_ld_b128(lds_off(Bs + r * GP + c),
                    Vb + (size_t)(n0 + r) * 2048 + k0 + c);
      async_ld_b128(lds_off(Bs + r * GP + c + 8),
                    Vb + (size_t)(n0 + r) * 2048 + k0 + c + 8);
    }
  };

  v8f acc[2] = {};
  const int kend = m0 + 32;      // causal: P defined (and nonzero) only for k < m0+32
  stage(0, dynsmem);
  int cur = 0;
  for (int k0 = 0; k0 < kend; k0 += 32) {
    if (k0 + 32 < kend) {
      stage(k0 + 32, dynsmem + (cur ^ 1) * OUT_STRIDE);
      wait_async<3>();
    } else {
      wait_async<0>();
    }
    __syncthreads();
    const u16* As = dynsmem + cur * OUT_STRIDE;
    const u16* Bs = As + 32 * GP;
    v16bf a  = frag_lds(As + (wm * 16) * GP, GP, lane);
    v16bf b0 = frag_lds(Bs + (wn * 32 +  0) * GP, GP, lane);
    v16bf b1 = frag_lds(Bs + (wn * 32 + 16) * GP, GP, lane);
    acc[0] = wmma_bf16(a, b0, acc[0]);
    acc[1] = wmma_bf16(a, b1, acc[1]);
    __syncthreads();
    cur ^= 1;
  }
  const int hi = (lane >> 4) << 3;
  const int nl = lane & 15;
  const int mt = m0 + wm * 16 + hi;
#pragma unroll
  for (int j = 0; j < 2; ++j) {
    const int nt = n0 + wn * 32 + j * 16 + nl;
#pragma unroll
    for (int r = 0; r < 8; ++r) {
      const float rl = 1.0f / Lrow[b * 2048 + mt + r];
      Out[((size_t)b * 2048 + mt + r) * 1024 + nt] = acc[j][r] * rl;
    }
  }
}

// ---------- host-side launch ----------
extern "C" void kernel_launch(void* const* d_in, const int* in_sizes, int n_in,
                              void* d_out, int out_size, void* d_ws, size_t ws_size,
                              hipStream_t stream) {
  const float* emb = (const float*)d_in[0];   // [4,2048,1024]
  const float* Wk  = (const float*)d_in[1];   // [1024,1024]
  const float* Wq  = (const float*)d_in[2];
  const float* Wv  = (const float*)d_in[3];
  float* out = (float*)d_out;

  char* ws = (char*)d_ws;
  // workspace layout (bytes)
  u16*   Xbf  = (u16*)(ws + 0);                       // 16 MiB
  u16*   Qbf  = (u16*)(ws + (size_t)16777216);        // 16 MiB
  u16*   Kbf  = (u16*)(ws + (size_t)33554432);        // 16 MiB
  u16*   Vtbf = (u16*)(ws + (size_t)50331648);        // 16 MiB (transposed V)
  u16*   Pbuf = (u16*)(ws + (size_t)67108864);        // 32 MiB
  float* Lrow = (float*)(ws + (size_t)100663296);     // 32 KiB
  u16*   Wqb  = (u16*)(ws + (size_t)100696064);       // 2 MiB
  u16*   Wkb  = (u16*)(ws + (size_t)102793216);       // 2 MiB
  u16*   Wvb  = (u16*)(ws + (size_t)104890368);       // 2 MiB

  // 1) convert fp32 -> bf16
  cvt_f32_bf16<<<8192, 256, 0, stream>>>(emb, Xbf, 2097152);
  cvt_f32_bf16<<<1024, 256, 0, stream>>>(Wq, Wqb, 262144);
  cvt_f32_bf16<<<1024, 256, 0, stream>>>(Wk, Wkb, 262144);
  cvt_f32_bf16<<<1024, 256, 0, stream>>>(Wv, Wvb, 262144);

  // 2) fused QKV projection (z = 0:Q, 1:K, 2:V-transposed), double-buffered
  proj_gemm<<<dim3(8, 128, 3), 256, 2 * PROJ_STRIDE * sizeof(u16), stream>>>(
      Xbf, Wqb, Wkb, Wvb, Qbf, Kbf, Vtbf);

  // 3) causal scores + softmax (Q tile resident in WGP LDS, async staging)
  attn_scores<<<dim3(64, 4), 64, 64 * QP * sizeof(u16) + 32 * sizeof(float),
                stream>>>(Qbf, Kbf, Pbuf, Lrow);

  // 4) O = (P @ V) / L, double-buffered
  attn_out<<<dim3(8, 64, 4), 256, 2 * OUT_STRIDE * sizeof(u16), stream>>>(
      Pbuf, Vtbf, Lrow, out);
}
